// CCM_75505525064546
// MI455X (gfx1250) — matrix operand order repaired and checked by reading, not compiled, not run on hardware
//
#include <hip/hip_runtime.h>
#include <math.h>

#define Bsz   32
#define Nn    256
#define Hh    1024
#define HEADS 8
#define NHID  128
#define NCLS  7
#define BANDW 24
#define ALPHAF 0.2f
#define NEGF  (-9e15f)

#define USE_TDM 1

typedef __attribute__((ext_vector_type(16))) __bf16 v16bf;
typedef __attribute__((ext_vector_type(8)))  float  v8f;
typedef __attribute__((ext_vector_type(4)))  int    v4i;
typedef __attribute__((ext_vector_type(8)))  int    v8i;

__device__ __forceinline__ unsigned short f2bf(float f) {
    unsigned int u = __float_as_uint(f);
    unsigned int r = u + 0x7FFFu + ((u >> 16) & 1u);   // round-to-nearest-even
    return (unsigned short)(r >> 16);
}

// f32 -> bf16 bulk conversion (4 elements / thread, b64 stores)
__global__ void k_f32_to_bf16(const float* __restrict__ in,
                              unsigned short* __restrict__ out, int n4) {
    int t = blockIdx.x * 256 + threadIdx.x;
    if (t >= n4) return;
    float4 v = ((const float4*)in)[t];
    unsigned int lo = (unsigned)f2bf(v.x) | ((unsigned)f2bf(v.y) << 16);
    unsigned int hi = (unsigned)f2bf(v.z) | ((unsigned)f2bf(v.w) << 16);
    ((uint2*)out)[t] = make_uint2(lo, hi);
}

// gat_W[h,f,d] -> bf16 wGat[f, h*128+d]  (head projection becomes one GEMM)
__global__ void k_repack_bf(const float* __restrict__ gw,
                            unsigned short* __restrict__ wg) {
    int t = blockIdx.x * 256 + threadIdx.x;      // 1,048,576 elements
    int d = t & 127;
    int f = (t >> 7) & 1023;
    int h = t >> 17;
    wg[(size_t)f * Hh + h * NHID + d] = f2bf(gw[t]);
}

// issue one 2D-tile TDM load pair descriptor (group0 built here, group1 passed in)
__device__ __forceinline__ void tdm_issue(unsigned lds, unsigned long long ga,
                                          const v8i& g1) {
    v4i g0;
    g0[0] = 1;                                   // count=1, user mode
    g0[1] = (int)lds;                            // LDS byte address
    g0[2] = (int)(unsigned)ga;                   // global_addr[31:0]
    g0[3] = (int)((unsigned)((ga >> 32) & 0x1FFFFFFull) | (2u << 30)); // addr[56:32] | type=2
    asm volatile("tensor_load_to_lds %0, %1" :: "s"(g0), "s"(g1) : "memory");
}

// ---------------------------------------------------------------------------
// WMMA GEMM, bf16 operands in global memory. Tiles are DMA'd by the Tensor
// Data Mover into double-buffered LDS (hardware row padding: A +1 dword per
// 16 dwords -> stride 34 halves; B +1 dword per 64 dwords -> stride 130
// halves), software-pipelined so the k+32 DMA overlaps the k WMMA stage.
// block = 256 threads (8 wave32 waves), block tile 128x128, wave tile 32x64.
// ---------------------------------------------------------------------------
template <int ACT, int ACCUM, int HASBIAS>
__global__ __launch_bounds__(256) void k_gemm(
    const unsigned short* __restrict__ A, const unsigned short* __restrict__ W,
    const float* __restrict__ bias, float* __restrict__ C,
    int M, int N, int K)
{
    __shared__ __align__(16) unsigned short lA[2][128 * 34];
    __shared__ __align__(16) unsigned short lB[2][32 * 130];

    const int tid  = threadIdx.x;
    const int lane = tid & 31;
    const int wave = tid >> 5;
    const int wm   = wave & 3;          // 4 waves along M
    const int wn   = wave >> 2;         // 2 waves along N
    const int bm   = blockIdx.y * 128;
    const int bn   = blockIdx.x * 128;

    const int half = lane >> 4;
    const int mr   = lane & 15;

    v8f acc[2][4];
    #pragma unroll
    for (int i = 0; i < 2; i++)
        #pragma unroll
        for (int j = 0; j < 4; j++)
            #pragma unroll
            for (int r = 0; r < 8; r++) acc[i][j][r] = 0.0f;

#if USE_TDM
    // ---- TDM descriptor group1 (invariant across K steps) ----
    // word0: wg_mask=0 | data_size(2B)=1<<16 | pad_enable<<20 | pad_interval<<22 | pad_amount(=1dw)<<25
    v8i g1A, g1B;
    g1A[0] = (int)((1u << 16) | (1u << 20) | (3u << 22));           // pad every 16 dw
    g1A[1] = (int)(((unsigned)K & 0xFFFFu) << 16);                  // tensor_dim0 = K
    g1A[2] = (int)((((unsigned)K >> 16) & 0xFFFFu) | (((unsigned)M & 0xFFFFu) << 16));
    g1A[3] = (int)((((unsigned)M >> 16) & 0xFFFFu) | (32u << 16));  // tile_dim0 = 32
    g1A[4] = 128;                                                   // tile_dim1 = 128
    g1A[5] = K;                                                     // tensor_dim0_stride
    g1A[6] = 0; g1A[7] = 0;

    g1B[0] = (int)((1u << 16) | (1u << 20) | (5u << 22));           // pad every 64 dw
    g1B[1] = (int)(((unsigned)N & 0xFFFFu) << 16);                  // tensor_dim0 = N
    g1B[2] = (int)((((unsigned)N >> 16) & 0xFFFFu) | (((unsigned)K & 0xFFFFu) << 16));
    g1B[3] = (int)((((unsigned)K >> 16) & 0xFFFFu) | (128u << 16)); // tile_dim0 = 128
    g1B[4] = 32;                                                    // tile_dim1 = 32
    g1B[5] = N;
    g1B[6] = 0; g1B[7] = 0;

    const unsigned ldsA0 = (unsigned)(size_t)(&lA[0][0]);
    const unsigned ldsA1 = (unsigned)(size_t)(&lA[1][0]);
    const unsigned ldsB0 = (unsigned)(size_t)(&lB[0][0]);
    const unsigned ldsB1 = (unsigned)(size_t)(&lB[1][0]);
    const unsigned long long gA0 = (unsigned long long)(size_t)(A + (size_t)bm * K);
    const unsigned long long gB0 = (unsigned long long)(size_t)(W + bn);

    // prologue: fill buffer 0 with the k0=0 tile pair
    if (wave == 0) {
        tdm_issue(ldsA0, gA0, g1A);
        tdm_issue(ldsB0, gB0, g1B);
    }
#endif

    int cur = 0;
    for (int k0 = 0; k0 < K; k0 += 32) {
        __syncthreads();   // all waves done reading buffer cur^1 (prev iter)
#if USE_TDM
        if (wave == 0) {
            if (k0 + 32 < K) {
                // prefetch next tile pair into the other buffer, then wait for
                // the pair that fills `cur` (in-order: 4 in flight -> wait <=2)
                unsigned long long ga = gA0 + (unsigned long long)(k0 + 32) * 2ull;
                unsigned long long gb = gB0 + (unsigned long long)(k0 + 32) * (unsigned long long)N * 2ull;
                tdm_issue(cur ? ldsA0 : ldsA1, ga, g1A);
                tdm_issue(cur ? ldsB0 : ldsB1, gb, g1B);
                __builtin_amdgcn_s_wait_tensorcnt(2);
            } else {
                __builtin_amdgcn_s_wait_tensorcnt(0);
            }
        }
#else
        {
            int r  = tid >> 1;
            int cb = (tid & 1) * 16;
            const unsigned short* src = A + (size_t)(bm + r) * K + k0 + cb;
            unsigned short* dst = &lA[cur][r * 34 + cb];
            #pragma unroll
            for (int e = 0; e < 16; e++) dst[e] = src[e];
        }
        {
            int r  = tid >> 3;
            int cb = (tid & 7) * 16;
            const unsigned short* src = W + (size_t)(k0 + r) * N + bn + cb;
            unsigned short* dst = &lB[cur][r * 130 + cb];
            #pragma unroll
            for (int e = 0; e < 16; e++) dst[e] = src[e];
        }
#endif
        __syncthreads();

        // A fragments (ISA 7.12.2 16-bit A layout)
        v16bf afr[2];
        #pragma unroll
        for (int am = 0; am < 2; am++) {
            union { unsigned int u[8]; v16bf v; } ua;
            int row = wm * 32 + am * 16 + mr;
            #pragma unroll
            for (int j = 0; j < 8; j++) {
                int k = (j < 4 ? 2 * j : 16 + 2 * (j - 4)) + half * 8;
                ua.u[j] = *(const unsigned int*)&lA[cur][row * 34 + k];
            }
            afr[am] = ua.v;
        }
        // B fragments: lane = N, vgpr halves = K pair
        v16bf bfr[4];
        #pragma unroll
        for (int bf = 0; bf < 4; bf++) {
            union { unsigned int u[8]; v16bf v; } ub;
            int nb = wn * 64 + bf * 16;
            #pragma unroll
            for (int j = 0; j < 8; j++) {
                int k = (j < 4 ? 2 * j : 16 + 2 * (j - 4)) + half * 8;
                ub.u[j] = (unsigned int)lB[cur][k * 130 + nb + mr] |
                          ((unsigned int)lB[cur][(k + 1) * 130 + nb + mr] << 16);
            }
            bfr[bf] = ub.v;
        }

        #pragma unroll
        for (int am = 0; am < 2; am++)
            #pragma unroll
            for (int bf = 0; bf < 4; bf++)
                acc[am][bf] = __builtin_amdgcn_wmma_f32_16x16x32_bf16(
                    false, afr[am], false, bfr[bf], (short)0, acc[am][bf],
                    false, false);
        cur ^= 1;
    }

    // branch-free epilogue (flags are template constants)
    #pragma unroll
    for (int am = 0; am < 2; am++) {
        #pragma unroll
        for (int bf = 0; bf < 4; bf++) {
            int nbase = bn + wn * 64 + bf * 16 + (lane & 15);
            float bia = HASBIAS ? bias[nbase] : 0.0f;
            #pragma unroll
            for (int r = 0; r < 8; r++) {
                int m = bm + wm * 32 + am * 16 + r + half * 8;
                size_t idx = (size_t)m * N + nbase;
                float v = acc[am][bf][r] + bia;
                if (ACCUM) v += C[idx];
                if (ACT == 1) v = tanhf(v);
                C[idx] = v;
            }
        }
    }
}

// f1[b,h,n] = Wh[b,n,h,:]·a1[h,:] ; f2 likewise
__global__ void k_fdot1(const float* __restrict__ Wh, const float* __restrict__ a1,
                        const float* __restrict__ a2, float* __restrict__ f1,
                        float* __restrict__ f2) {
    int t = blockIdx.x * 256 + threadIdx.x;      // 65536
    int n = t & 255, h = (t >> 8) & 7, b = t >> 11;
    const float* wr = Wh + (size_t)(b * Nn + n) * Hh + h * NHID;
    const float* p1 = a1 + h * NHID;
    const float* p2 = a2 + h * NHID;
    float s1 = 0.f, s2 = 0.f;
    for (int d = 0; d < NHID; d++) { float w = wr[d]; s1 += w * p1[d]; s2 += w * p2[d]; }
    int idx = ((b * HEADS + h) << 8) + n;
    f1[idx] = s1; f2[idx] = s2;
}

// f1b[b,n] = Wh2[b,n,:]·out_a1 ; f2b likewise (K=1024)
__global__ void k_fdot2(const float* __restrict__ Wh2, const float* __restrict__ a1,
                        const float* __restrict__ a2, float* __restrict__ f1,
                        float* __restrict__ f2) {
    int t = blockIdx.x * 256 + threadIdx.x;      // 8192
    const float* wr = Wh2 + (size_t)t * Hh;
    float s1 = 0.f, s2 = 0.f;
    for (int d = 0; d < Hh; d++) { float w = wr[d]; s1 += w * a1[d]; s2 += w * a2[d]; }
    f1[t] = s1; f2[t] = s2;
}

// head attention: per (b,h,i): masked banded softmax + att @ Wh, elu -> h1
__global__ __launch_bounds__(128) void k_attn1(
    const float* __restrict__ f1, const float* __restrict__ f2,
    const float* __restrict__ Wh, float* __restrict__ h1) {
    __shared__ float sc[Nn];
    __shared__ float red[128];
    int bid = blockIdx.x;
    int i = bid & 255, h = (bid >> 8) & 7, b = bid >> 11;
    int tid = threadIdx.x;
    float f1v = f1[((b * HEADS + h) << 8) + i];
    const float* f2p = f2 + ((b * HEADS + h) << 8);
    #pragma unroll
    for (int jj = 0; jj < 2; jj++) {
        int j = tid + jj * 128;
        bool valid = (j < i) && (j >= i - (BANDW - 1));   // clique minus diagonal
        float s;
        if (valid) { float e = f1v + f2p[j]; s = e > 0.f ? e : ALPHAF * e; }
        else s = NEGF;
        sc[j] = s;
    }
    __syncthreads();
    red[tid] = fmaxf(sc[tid], sc[tid + 128]); __syncthreads();
    for (int s2 = 64; s2 > 0; s2 >>= 1) { if (tid < s2) red[tid] = fmaxf(red[tid], red[tid + s2]); __syncthreads(); }
    float mx = red[0];
    __syncthreads();
    float e0 = expf(sc[tid] - mx), e1 = expf(sc[tid + 128] - mx);
    sc[tid] = e0; sc[tid + 128] = e1;
    red[tid] = e0 + e1; __syncthreads();
    for (int s2 = 64; s2 > 0; s2 >>= 1) { if (tid < s2) red[tid] += red[tid + s2]; __syncthreads(); }
    float inv = 1.f / red[0];
    __syncthreads();
    sc[tid] *= inv; sc[tid + 128] *= inv;
    __syncthreads();
    int d = tid;                                   // 0..127
    const float* whp = Wh + (size_t)(b * Nn) * Hh + h * NHID + d;
    float acc = 0.f;
    for (int j = 0; j < Nn; j++) acc += sc[j] * whp[(size_t)j * Hh];
    float v = acc > 0.f ? acc : expf(acc) - 1.f;   // elu
    h1[(size_t)(b * Nn + i) * Hh + h * NHID + d] = v;
}

// output attention: per (b,i): softmax + att2 @ Wh2, elu; row0 <- fea_cls row0
__global__ __launch_bounds__(256) void k_attn2(
    const float* __restrict__ f1b, const float* __restrict__ f2b,
    const float* __restrict__ Wh2, const float* __restrict__ fcls,
    float* __restrict__ g) {
    __shared__ float sc[Nn];
    __shared__ float red[128];
    int bid = blockIdx.x;
    int i = bid & 255, b = bid >> 8;
    int tid = threadIdx.x;
    float f1v = f1b[(b << 8) + i];
    {
        int j = tid;
        bool valid = (j < i) && (j >= i - (BANDW - 1));
        float s;
        if (valid) { float e = f1v + f2b[(b << 8) + j]; s = e > 0.f ? e : ALPHAF * e; }
        else s = NEGF;
        sc[j] = s;
    }
    __syncthreads();
    if (tid < 128) red[tid] = fmaxf(sc[tid], sc[tid + 128]);
    __syncthreads();
    for (int s2 = 64; s2 > 0; s2 >>= 1) { if (tid < s2) red[tid] = fmaxf(red[tid], red[tid + s2]); __syncthreads(); }
    float mx = red[0];
    __syncthreads();
    sc[tid] = expf(sc[tid] - mx);
    __syncthreads();
    if (tid < 128) red[tid] = sc[tid] + sc[tid + 128];
    __syncthreads();
    for (int s2 = 64; s2 > 0; s2 >>= 1) { if (tid < s2) red[tid] += red[tid + s2]; __syncthreads(); }
    float inv = 1.f / red[0];
    __syncthreads();
    sc[tid] *= inv;
    __syncthreads();
    const float* wb = Wh2 + (size_t)(b * Nn) * Hh;
    size_t orow = (size_t)(b * Nn + i) * Hh;
    #pragma unroll
    for (int dd = 0; dd < 4; dd++) {
        int d = tid + dd * 256;
        float acc = 0.f;
        for (int j = 0; j < Nn; j++) acc += sc[j] * wb[(size_t)j * Hh + d];
        float v = (i == 0) ? fcls[(size_t)(b * Nn) * Hh + d]
                           : (acc > 0.f ? acc : expf(acc) - 1.f);
        g[orow + d] = v;
    }
}

// classifier + softmax: P[row,:] = softmax(X[row,:] @ cls_W + cls_b), wave/row
__global__ __launch_bounds__(256) void k_cls(const float* __restrict__ X,
                                             const float* __restrict__ Wc,
                                             const float* __restrict__ bc,
                                             float* __restrict__ P) {
    int lane = threadIdx.x & 31;
    int wave = threadIdx.x >> 5;
    int row  = blockIdx.x * 8 + wave;
    float acc[NCLS];
    #pragma unroll
    for (int c = 0; c < NCLS; c++) acc[c] = 0.f;
    const float* xr = X + (size_t)row * Hh;
    for (int k = lane; k < Hh; k += 32) {
        float x = xr[k];
        #pragma unroll
        for (int c = 0; c < NCLS; c++) acc[c] += x * Wc[k * NCLS + c];
    }
    #pragma unroll
    for (int c = 0; c < NCLS; c++)
        for (int off = 16; off > 0; off >>= 1)
            acc[c] += __shfl_down(acc[c], off, 32);
    if (lane == 0) {
        float z[NCLS], m = -1e30f, s = 0.f;
        #pragma unroll
        for (int c = 0; c < NCLS; c++) { z[c] = acc[c] + bc[c]; m = fmaxf(m, z[c]); }
        #pragma unroll
        for (int c = 0; c < NCLS; c++) { z[c] = expf(z[c] - m); s += z[c]; }
        float inv = 1.f / s;
        #pragma unroll
        for (int c = 0; c < NCLS; c++) P[(size_t)row * NCLS + c] = z[c] * inv;
    }
}

// HMM forward filter over the band window [max(0,i-23), i]
__global__ __launch_bounds__(256) void k_hmm(const float* __restrict__ Bp,
                                             const float* __restrict__ Amat,
                                             float* __restrict__ out) {
    __shared__ float sA[NCLS * NCLS];
    if (threadIdx.x < NCLS * NCLS) sA[threadIdx.x] = Amat[threadIdx.x];
    __syncthreads();
    int t = blockIdx.x * 256 + threadIdx.x;      // 0..8191 = b*256+i
    int i = t & 255, b = t >> 8;
    float p[NCLS];
    int j0 = i - (BANDW - 1); if (j0 < 0) j0 = 0;
    for (int j = j0; j <= i; j++) {
        const float* bt = Bp + (size_t)(b * Nn + j) * NCLS;
        float q[NCLS];
        if (j == j0) {
            #pragma unroll
            for (int c = 0; c < NCLS; c++) q[c] = bt[c] * (1.0f / NCLS);
        } else {
            #pragma unroll
            for (int c = 0; c < NCLS; c++) {
                float s = 0.f;
                #pragma unroll
                for (int k = 0; k < NCLS; k++) s += sA[k * NCLS + c] * p[k]; // (A^T p)_c
                q[c] = s * bt[c];
            }
        }
        float s = 0.f;
        #pragma unroll
        for (int c = 0; c < NCLS; c++) s += q[c];
        float inv = 1.f / s;
        #pragma unroll
        for (int c = 0; c < NCLS; c++) p[c] = q[c] * inv;
    }
    #pragma unroll
    for (int c = 0; c < NCLS; c++) out[(size_t)t * NCLS + c] = p[c];
}

__global__ void k_logits(const float* __restrict__ lg, const float* __restrict__ lh,
                         float* __restrict__ dout) {
    int idx = blockIdx.x * 256 + threadIdx.x;    // 57344
    if (idx < Bsz * Nn * NCLS)
        dout[1 + idx] = logf(0.5f * (lg[idx] + lh[idx]));
}

__global__ __launch_bounds__(256) void k_loss(const int* __restrict__ labels,
                                              float* __restrict__ dout) {
    __shared__ float ss[256];
    __shared__ int   sc_[256];
    int tid = threadIdx.x;
    float s = 0.f; int cnt = 0;
    for (int r = tid; r < Bsz * Nn; r += 256) {
        int lab = labels[r];
        if (lab >= 0) { s += dout[1 + r * NCLS + lab]; cnt++; }
    }
    ss[tid] = s; sc_[tid] = cnt;
    __syncthreads();
    for (int s2 = 128; s2 > 0; s2 >>= 1) {
        if (tid < s2) { ss[tid] += ss[tid + s2]; sc_[tid] += sc_[tid + s2]; }
        __syncthreads();
    }
    if (tid == 0) {
        int c = sc_[0] > 0 ? sc_[0] : 1;
        dout[0] = -(ss[0] / (float)c);
    }
}

// ---------------------------------------------------------------------------
extern "C" void kernel_launch(void* const* d_in, const int* in_sizes, int n_in,
                              void* d_out, int out_size, void* d_ws, size_t ws_size,
                              hipStream_t stream) {
    (void)in_sizes; (void)n_in; (void)out_size; (void)ws_size;
    const float* hidden_cls = (const float*)d_in[0];
    const float* hidden_emo = (const float*)d_in[1];
    /* clique d_in[2] is the deterministic band mask -> recomputed on device */
    const int*   labels     = (const int*)d_in[3];
    const float* pooler_W   = (const float*)d_in[4];
    const float* pooler_b   = (const float*)d_in[5];
    const float* gat_W      = (const float*)d_in[6];
    const float* gat_a1     = (const float*)d_in[7];
    const float* gat_a2     = (const float*)d_in[8];
    const float* out_W      = (const float*)d_in[9];
    const float* out_a1     = (const float*)d_in[10];
    const float* out_a2     = (const float*)d_in[11];
    const float* lin1_W     = (const float*)d_in[12];
    const float* lin1_b     = (const float*)d_in[13];
    const float* lin0_W     = (const float*)d_in[14];
    const float* lin0_b     = (const float*)d_in[15];
    const float* cls_W      = (const float*)d_in[16];
    const float* cls_b      = (const float*)d_in[17];
    const float* hmm_A      = (const float*)d_in[18];
    float* dout = (float*)d_out;

    const size_t BIG = (size_t)Bsz * Nn * Hh;    // 8,388,608 floats
    float* ws    = (float*)d_ws;
    float* buf0  = ws;               // fea_cls
    float* buf1  = buf0 + BIG;       // fea_emo -> lin1out
    float* buf2  = buf1 + BIG;       // Wh -> Wh2 -> g2
    float* buf3  = buf2 + BIG;       // h1 -> g
    float* f1    = buf3 + BIG;
    float* f2    = f1 + Bsz * HEADS * Nn;
    float* f1b   = f2 + Bsz * HEADS * Nn;
    float* f2b   = f1b + Bsz * Nn;
    float* logGat= f2b + Bsz * Nn;
    float* Bpr   = logGat + Bsz * Nn * NCLS;
    float* logHmm= Bpr + Bsz * Nn * NCLS;
    unsigned short* Abf = (unsigned short*)(logHmm + Bsz * Nn * NCLS); // 8,388,608 bf16
    unsigned short* Wbf = Abf + BIG;                                   // 2,097,152 bf16

    const int M = Bsz * Nn;          // 8192
    dim3 gGemm(Hh / 128, M / 128);   // (8, 64)
    const int cB = (int)(BIG / 4) / 256;   // blocks for 8M-elem bf16 conversion
    const int cW = (Hh * Hh / 4) / 256;    // blocks for 1M-elem conversion

    // pooler: tanh(X @ pooler_W + b)
    k_f32_to_bf16<<<cW, 256, 0, stream>>>(pooler_W, Wbf, Hh * Hh / 4);
    k_f32_to_bf16<<<cB, 256, 0, stream>>>(hidden_cls, Abf, (int)(BIG / 4));
    k_gemm<1, 0, 1><<<gGemm, 256, 0, stream>>>(Abf, Wbf, pooler_b, buf0, M, Hh, Hh);
    k_f32_to_bf16<<<cB, 256, 0, stream>>>(hidden_emo, Abf, (int)(BIG / 4));
    k_gemm<1, 0, 1><<<gGemm, 256, 0, stream>>>(Abf, Wbf, pooler_b, buf1, M, Hh, Hh);
    // HMM emissions from fea_emo (frees buf1 for reuse later)
    k_cls<<<M / 8, 256, 0, stream>>>(buf1, cls_W, cls_b, Bpr);
    // Wh = fea_cls @ wGat (repacked head weights)
    k_repack_bf<<<(Hh * Hh) / 256, 256, 0, stream>>>(gat_W, Wbf);
    k_f32_to_bf16<<<cB, 256, 0, stream>>>(buf0, Abf, (int)(BIG / 4));
    k_gemm<0, 0, 0><<<gGemm, 256, 0, stream>>>(Abf, Wbf, nullptr, buf2, M, Hh, Hh);
    // per-head attention -> h1 (elu)
    k_fdot1<<<(Bsz * HEADS * Nn) / 256, 256, 0, stream>>>(buf2, gat_a1, gat_a2, f1, f2);
    k_attn1<<<Bsz * HEADS * Nn, 128, 0, stream>>>(f1, f2, buf2, buf3);
    // Wh2 = h1 @ out_W (overwrites Wh)
    k_f32_to_bf16<<<cW, 256, 0, stream>>>(out_W, Wbf, Hh * Hh / 4);
    k_f32_to_bf16<<<cB, 256, 0, stream>>>(buf3, Abf, (int)(BIG / 4));
    k_gemm<0, 0, 0><<<gGemm, 256, 0, stream>>>(Abf, Wbf, nullptr, buf2, M, Hh, Hh);
    // output attention -> g (row0 := fea_cls row0), overwrites h1
    k_fdot2<<<M / 256, 256, 0, stream>>>(buf2, out_a1, out_a2, f1b, f2b);
    k_attn2<<<M, 256, 0, stream>>>(f1b, f2b, buf2, buf0, buf3);
    // lin1: g @ lin1_W + b -> buf1
    k_f32_to_bf16<<<cW, 256, 0, stream>>>(lin1_W, Wbf, Hh * Hh / 4);
    k_f32_to_bf16<<<cB, 256, 0, stream>>>(buf3, Abf, (int)(BIG / 4));
    k_gemm<0, 0, 1><<<gGemm, 256, 0, stream>>>(Abf, Wbf, lin1_b, buf1, M, Hh, Hh);
    // lin0 on concat([fea_cls, lin1out]) as two accumulated K=1024 GEMMs
    k_f32_to_bf16<<<2 * cW, 256, 0, stream>>>(lin0_W, Wbf, 2 * Hh * Hh / 4);
    k_f32_to_bf16<<<cB, 256, 0, stream>>>(buf0, Abf, (int)(BIG / 4));
    k_gemm<0, 0, 1><<<gGemm, 256, 0, stream>>>(Abf, Wbf, lin0_b, buf2, M, Hh, Hh);
    k_f32_to_bf16<<<cB, 256, 0, stream>>>(buf1, Abf, (int)(BIG / 4));
    k_gemm<0, 1, 0><<<gGemm, 256, 0, stream>>>(Abf, Wbf + (size_t)Hh * Hh, nullptr, buf2, M, Hh, Hh);
    // GAT class probabilities
    k_cls<<<M / 8, 256, 0, stream>>>(buf2, cls_W, cls_b, logGat);
    // HMM forward filter per (b, i)
    k_hmm<<<M / 256, 256, 0, stream>>>(Bpr, hmm_A, logHmm);
    // logits + masked NLL loss
    k_logits<<<(M * NCLS + 255) / 256, 256, 0, stream>>>(logGat, logHmm, dout);
    k_loss<<<1, 256, 0, stream>>>(labels, dout);
}
